// RefGatedMLPFusedMoE_14809047237035
// MI455X (gfx1250) — matrix operand order repaired and checked by reading, not compiled, not run on hardware
//
#include <hip/hip_runtime.h>
#include <hip/hip_bf16.h>

// ---------------------------------------------------------------------------
// Fused MoE gated MLP for MI455X (gfx1250), bf16 WMMA path.
// T=8192 tokens, H=1024, F=2816, E=8 experts, top-2 routing.
// Preferred path: weights pre-converted to bf16 in workspace (fits in 192MB L2),
// GEMM reads bf16 directly into WMMA operands. Fallback: fp32->bf16 in-kernel.
// ---------------------------------------------------------------------------

typedef __bf16 v16bf __attribute__((ext_vector_type(16)));
typedef __bf16 v8bf  __attribute__((ext_vector_type(8)));
typedef float  v8f   __attribute__((ext_vector_type(8)));

static constexpr int T = 8192;
static constexpr int H = 1024;
static constexpr int F = 2816;
static constexpr int E = 8;
static constexpr int BM = 16;        // tokens per block tile
static constexpr int FC = 128;       // F chunk per block iteration (8 waves * 16)
static constexpr int XPITCH = 24;    // xT sized H*XPITCH halfs; rows indexed [tok][h], pitch H
static constexpr int YPITCH = 24;    // yT sized FC*YPITCH halfs; rows indexed [tok][f], pitch FC

__device__ __forceinline__ v8f wmma_bf16(v16bf a, v16bf b, v8f c) {
  return __builtin_amdgcn_wmma_f32_16x16x32_bf16(
      /*neg_a=*/false, a, /*neg_b=*/false, b,
      /*c_mod=*/(short)0, c, /*reuse_a=*/false, /*reuse_b=*/false);
}

// fp32 weight row -> bf16 A operand (two 8-elem runs at base, base+16)
__device__ __forceinline__ v16bf load_w16_f32(const float* __restrict__ row, int base) {
  const float4* p = (const float4*)(row + base);
  const float4* q = (const float4*)(row + base + 16);
  float4 a = p[0], b = p[1], c = q[0], d = q[1];
  v16bf r;
  r[0]  = (__bf16)a.x; r[1]  = (__bf16)a.y; r[2]  = (__bf16)a.z; r[3]  = (__bf16)a.w;
  r[4]  = (__bf16)b.x; r[5]  = (__bf16)b.y; r[6]  = (__bf16)b.z; r[7]  = (__bf16)b.w;
  r[8]  = (__bf16)c.x; r[9]  = (__bf16)c.y; r[10] = (__bf16)c.z; r[11] = (__bf16)c.w;
  r[12] = (__bf16)d.x; r[13] = (__bf16)d.y; r[14] = (__bf16)d.z; r[15] = (__bf16)d.w;
  return r;
}

// bf16 weight row -> A operand, straight b128 loads, no conversion VALU
__device__ __forceinline__ v16bf load_w16_bf16(const __bf16* __restrict__ row, int base) {
  v8bf lo = *(const v8bf*)(row + base);
  v8bf hi = *(const v8bf*)(row + base + 16);
  return __builtin_shufflevector(lo, hi, 0,1,2,3,4,5,6,7,8,9,10,11,12,13,14,15);
}

// 16 contiguous bf16 from an LDS row (16B aligned) as the B operand.
__device__ __forceinline__ v16bf load_lds16(const __bf16* p) {
  const v8bf* q = (const v8bf*)p;
  v8bf lo = q[0], hi = q[1];
  return __builtin_shufflevector(lo, hi, 0,1,2,3,4,5,6,7,8,9,10,11,12,13,14,15);
}

// silu via hardware reciprocal (v_rcp_f32) -- avoids IEEE div macro expansion;
// rcp error (~1 ulp) is far below the bf16 quantization already in the matmuls.
__device__ __forceinline__ float fast_silu(float g) {
  return g * __builtin_amdgcn_rcpf(1.0f + __expf(-g));
}

// ---------------------------------------------------------------------------
// 1) zero output
// ---------------------------------------------------------------------------
__global__ void zero_kernel(float4* __restrict__ p) {
  int i = blockIdx.x * blockDim.x + threadIdx.x;   // T*H/4 float4
  p[i] = make_float4(0.f, 0.f, 0.f, 0.f);
}

// ---------------------------------------------------------------------------
// 1b) fp32 -> bf16 weight conversion (8 elements / thread)
// ---------------------------------------------------------------------------
__global__ void convert_bf16_kernel(const float* __restrict__ src,
                                    __bf16* __restrict__ dst, int n8) {
  int i = blockIdx.x * blockDim.x + threadIdx.x;
  if (i >= n8) return;
  const float4* s = (const float4*)src;
  float4 a = s[2 * i], b = s[2 * i + 1];
  v8bf r;
  r[0] = (__bf16)a.x; r[1] = (__bf16)a.y; r[2] = (__bf16)a.z; r[3] = (__bf16)a.w;
  r[4] = (__bf16)b.x; r[5] = (__bf16)b.y; r[6] = (__bf16)b.z; r[7] = (__bf16)b.w;
  ((v8bf*)dst)[i] = r;
}

// ---------------------------------------------------------------------------
// 2) router: top-2 of 8 logits, renormalized (softmax denominator cancels)
// ---------------------------------------------------------------------------
__global__ void router_kernel(const float* __restrict__ logits,
                              int* __restrict__ ti, float* __restrict__ tw) {
  int t = blockIdx.x * blockDim.x + threadIdx.x;
  if (t >= T) return;
  float l[E];
  #pragma unroll
  for (int i = 0; i < E; ++i) l[i] = logits[t * E + i];
  int i0 = 0; float b0 = l[0];
  #pragma unroll
  for (int i = 1; i < E; ++i) if (l[i] > b0) { b0 = l[i]; i0 = i; }
  int i1 = (i0 == 0) ? 1 : 0; float b1 = l[i1];
  #pragma unroll
  for (int i = 0; i < E; ++i) if (i != i0 && l[i] > b1) { b1 = l[i]; i1 = i; }
  float w0 = 1.0f / (1.0f + __expf(b1 - b0));      // = p0/(p0+p1)
  ti[2 * t + 0] = i0; ti[2 * t + 1] = i1;
  tw[2 * t + 0] = w0; tw[2 * t + 1] = 1.0f - w0;
}

// ---------------------------------------------------------------------------
// 3) compaction: one block per expert builds an ordered token list
// ---------------------------------------------------------------------------
__global__ void compact_kernel(const int* __restrict__ ti, const float* __restrict__ tw,
                               int* __restrict__ tokList, float* __restrict__ wtList,
                               int* __restrict__ counts) {
  int e = blockIdx.x;
  __shared__ int waveSums[8];
  __shared__ int waveBase[8];
  __shared__ int blockBase;
  if (threadIdx.x == 0) blockBase = 0;
  __syncthreads();
  int lane = threadIdx.x & 31, wave = threadIdx.x >> 5;
  for (int base = 0; base < T; base += 256) {
    int t = base + threadIdx.x;
    int a0 = ti[2 * t], a1 = ti[2 * t + 1];
    bool flag = (a0 == e) || (a1 == e);
    float w = (a0 == e) ? tw[2 * t] : tw[2 * t + 1];
    unsigned mask32 = (unsigned)__ballot(flag);          // wave32
    int prefix = __popc(mask32 & ((1u << lane) - 1u));
    if (lane == 0) waveSums[wave] = __popc(mask32);
    __syncthreads();
    if (threadIdx.x == 0) {
      int acc = blockBase;
      #pragma unroll
      for (int wv = 0; wv < 8; ++wv) { waveBase[wv] = acc; acc += waveSums[wv]; }
      blockBase = acc;
    }
    __syncthreads();
    if (flag) {
      int idx = waveBase[wave] + prefix;
      tokList[e * T + idx] = t;
      wtList[e * T + idx] = w;
    }
    __syncthreads();
  }
  if (threadIdx.x == 0) counts[e] = blockBase;
}

// ---------------------------------------------------------------------------
// 4) fused gate/up/silu/down GEMM per (expert, 16-token tile)
//    Orientation: tokens on the N/lane axis everywhere.
//      gate/up : D[f, tok]  = w1/w3 tile (A) x xT (B)
//      down    : D[h, tok] += w2 tile (A)  x yT (B)
//    Templated on weight type: bf16 (preferred) or fp32 (fallback w/ cvt).
// ---------------------------------------------------------------------------
template <typename WT>
__device__ __forceinline__ v16bf load_wtile(const WT* row, int base);
template <>
__device__ __forceinline__ v16bf load_wtile<float>(const float* row, int base) {
  return load_w16_f32(row, base);
}
template <>
__device__ __forceinline__ v16bf load_wtile<__bf16>(const __bf16* row, int base) {
  return load_w16_bf16(row, base);
}

template <typename WT>
__global__ __launch_bounds__(256) void moe_kernel(
    const float* __restrict__ x,
    const WT* __restrict__ w1, const WT* __restrict__ w3,
    const WT* __restrict__ w2,
    const int* __restrict__ tokList, const float* __restrict__ wtList,
    const int* __restrict__ counts,
    float* __restrict__ out) {
  int e = blockIdx.y;
  int tile = blockIdx.x;
  int cnt = counts[e];
  if (tile * BM >= cnt) return;

  __shared__ __bf16 xT[H * XPITCH];   // [tok][h] activations (row pitch H), 48KB
  __shared__ __bf16 yT[FC * YPITCH];  // [tok][f_local] silu(g)*u chunk, 6KB
  __shared__ int   tokLds[BM];
  __shared__ float coeffLds[BM];

  int tid = threadIdx.x;
  int lane = tid & 31;
  int wave = tid >> 5;
  int fl = lane & 15;     // A: M-row within tile / D,B: N = token column
  int kh = lane >> 4;     // which K-half this lane holds for A

  if (tid < BM) {
    int idx = tile * BM + tid;
    int tok = (idx < cnt) ? tokList[e * T + idx] : -1;
    tokLds[tid] = tok;
    coeffLds[tid] = (idx < cnt) ? wtList[e * T + idx] : 0.0f;
  }
  __syncthreads();

  // Stage x tile (fp32 -> bf16, row-major [tok][h]) into LDS.
  for (int i = tid; i < BM * (H / 4); i += 256) {
    int m = i >> 8;          // token row (H/4 = 256 float4 per row)
    int h4 = i & 255;
    int tok = tokLds[m];
    float4 v = make_float4(0.f, 0.f, 0.f, 0.f);
    if (tok >= 0) v = ((const float4*)(x + (size_t)tok * H))[h4];
    int h = h4 * 4;
    xT[m * H + h + 0] = (__bf16)v.x;
    xT[m * H + h + 1] = (__bf16)v.y;
    xT[m * H + h + 2] = (__bf16)v.z;
    xT[m * H + h + 3] = (__bf16)v.w;
  }
  __syncthreads();

  // Persistent output accumulators: wave owns H columns [wave*128, wave*128+128)
  v8f acc[8];
  #pragma unroll
  for (int j = 0; j < 8; ++j) acc[j] = (v8f){0.f,0.f,0.f,0.f,0.f,0.f,0.f,0.f};

  for (int ic = 0; ic < F / FC; ++ic) {
    // ---- gate & up: this wave computes 16 f-rows of the chunk -------------
    int fbase = ic * FC + wave * 16;
    const WT* w1row = w1 + ((size_t)e * F + fbase + fl) * H;
    const WT* w3row = w3 + ((size_t)e * F + fbase + fl) * H;
    v8f g = (v8f){0.f,0.f,0.f,0.f,0.f,0.f,0.f,0.f};
    v8f u = (v8f){0.f,0.f,0.f,0.f,0.f,0.f,0.f,0.f};
    #pragma unroll 2
    for (int h0 = 0; h0 < H; h0 += 32) {
      v16bf a1 = load_wtile<WT>(w1row, h0 + kh * 8);
      v16bf a3 = load_wtile<WT>(w3row, h0 + kh * 8);
      v16bf b  = load_w16_bf16(&xT[fl * H], h0 + kh * 8);
      g = wmma_bf16(a1, b, g);
      u = wmma_bf16(a3, b, u);
    }
    // ---- fused silu(g)*u -> yT (bf16) ------------------------------------
    #pragma unroll
    for (int r = 0; r < 8; ++r) {
      float yv = fast_silu(g[r]) * u[r];
      int fr = wave * 16 + r + 8 * kh;    // f row within chunk (M of D)
      yT[fl * FC + fr] = (__bf16)yv;  // fl = token column (N of D)
    }
    __syncthreads();
    // ---- down-projection partial: acc[h, tok] += w2 x yT -----------------
    #pragma unroll
    for (int k0 = 0; k0 < FC; k0 += 32) {
      v16bf b = load_w16_bf16(&yT[fl * FC], k0 + kh * 8);
      #pragma unroll
      for (int j = 0; j < 8; ++j) {
        const WT* w2row = w2 + ((size_t)e * H + wave * 128 + j * 16 + fl) * F;
        v16bf a = load_wtile<WT>(w2row, ic * FC + k0 + kh * 8);
        acc[j] = wmma_bf16(a, b, acc[j]);
      }
    }
    __syncthreads();   // before yT is overwritten next chunk
  }

  // ---- scale by routing coefficient, accumulate into out -----------------
  int tok = tokLds[fl];            // N = token column of D tiles
  if (tok >= 0) {
    float c = coeffLds[fl];
    float* orow = out + (size_t)tok * H;
    #pragma unroll
    for (int j = 0; j < 8; ++j) {
      #pragma unroll
      for (int r = 0; r < 8; ++r) {
        int h = wave * 128 + j * 16 + r + 8 * kh;   // M row of D tile j
        atomicAdd(orow + h, acc[j][r] * c);
      }
    }
  }
}

// ---------------------------------------------------------------------------
// Host launch
// ---------------------------------------------------------------------------
extern "C" void kernel_launch(void* const* d_in, const int* in_sizes, int n_in,
                              void* d_out, int out_size, void* d_ws, size_t ws_size,
                              hipStream_t stream) {
  const float* x      = (const float*)d_in[0];  // [T,H]
  const float* logits = (const float*)d_in[1];  // [T,E]
  const float* w1     = (const float*)d_in[2];  // [E,F,H]
  const float* w3     = (const float*)d_in[3];  // [E,F,H]
  const float* w2     = (const float*)d_in[4];  // [E,H,F]
  float* out = (float*)d_out;

  char* ws = (char*)d_ws;
  int*   counts  = (int*)ws;                                   // 8 ints
  int*   ti      = (int*)(ws + 256);                           // 2*T ints
  float* tw      = (float*)(ws + 256 + 2 * T * 4);             // 2*T floats
  int*   tokList = (int*)(ws + 256 + 4 * T * 4);               // E*T ints
  float* wtList  = (float*)(ws + 256 + 4 * T * 4 + E * T * 4); // E*T floats

  constexpr size_t WELEMS = (size_t)E * F * H;         // per weight tensor
  constexpr size_t WBYTES = WELEMS * 2;                // bf16 bytes
  constexpr size_t WB_OFF = 1u << 20;                  // 1MB: routing scratch
  const bool use_bf16 = ws_size >= WB_OFF + 3 * WBYTES;

  zero_kernel<<<(T * H / 4) / 256, 256, 0, stream>>>((float4*)out);
  router_kernel<<<T / 256, 256, 0, stream>>>(logits, ti, tw);
  compact_kernel<<<E, 256, 0, stream>>>(ti, tw, tokList, wtList, counts);

  if (use_bf16) {
    __bf16* w1b = (__bf16*)(ws + WB_OFF);
    __bf16* w3b = (__bf16*)(ws + WB_OFF + WBYTES);
    __bf16* w2b = (__bf16*)(ws + WB_OFF + 2 * WBYTES);
    const int n8 = (int)(WELEMS / 8);
    const int cg = (n8 + 255) / 256;
    convert_bf16_kernel<<<cg, 256, 0, stream>>>(w1, w1b, n8);
    convert_bf16_kernel<<<cg, 256, 0, stream>>>(w3, w3b, n8);
    convert_bf16_kernel<<<cg, 256, 0, stream>>>(w2, w2b, n8);
    moe_kernel<__bf16><<<dim3(T / BM, E), 256, 0, stream>>>(
        x, w1b, w3b, w2b, tokList, wtList, counts, out);
  } else {
    moe_kernel<float><<<dim3(T / BM, E), 256, 0, stream>>>(
        x, w1, w3, w2, tokList, wtList, counts, out);
  }
}
